// Net_42777874268533
// MI455X (gfx1250) — matrix operand order repaired and checked by reading, not compiled, not run on hardware
//
#include <hip/hip_runtime.h>

// ---------------------------------------------------------------------------
// Problem constants (from reference)
// ---------------------------------------------------------------------------
#define NNODES 50000
#define NEDGES 600000
#define CIN    64
#define COUT   128
#define KMIX   5

typedef __bf16 bf16;
typedef __attribute__((ext_vector_type(16))) __bf16        v16bf;
typedef __attribute__((ext_vector_type(8)))  float         v8f;
typedef __attribute__((ext_vector_type(4)))  unsigned int  u32x4;

union FragBF { v16bf v; u32x4 u[2]; };

static inline int ceil_div(int a, int b) { return (a + b - 1) / b; }

// ---------------------------------------------------------------------------
// f32 -> bf16 convert
// ---------------------------------------------------------------------------
__global__ void cvt_bf16_kernel(const float* __restrict__ in, bf16* __restrict__ out, int n) {
    int i = blockIdx.x * blockDim.x + threadIdx.x;
    if (i < n) out[i] = (bf16)in[i];
}

__global__ void fill_zero_kernel(float* __restrict__ p, int n) {
    int i = blockIdx.x * blockDim.x + threadIdx.x;
    if (i < n) p[i] = 0.0f;
}

// ---------------------------------------------------------------------------
// WMMA bf16 GEMM: C[M x Ncols] = A[M x Kdim] * B[Kdim x Ncols] (+bias[col])
// Block = 128 threads = 4 waves; block tile 64x64; each wave does a 16x64
// strip via 4x v_wmma_f32_16x16x32_bf16. Kdim % 32 == 0, Ncols % 64 == 0.
// A tile is staged memory->LDS with GLOBAL_LOAD_ASYNC_TO_LDS_B128 (ASYNCcnt),
// no VGPR round trip; B tile is transposed in-flight with regular loads.
// Fragment layouts follow CDNA5 ISA 7.12.2 (16-bit A 16x32, B 32x16, f32 C).
// ---------------------------------------------------------------------------
#define TM 64
#define TN 64
#define TK 32
#define LPAD 40   // row pitch in bf16 elems: 80B, keeps 16B alignment, skews banks

__global__ __launch_bounds__(128)
void wmma_gemm_bf16_kernel(const bf16* __restrict__ A,
                           const bf16* __restrict__ B,
                           const float* __restrict__ bias,
                           float* __restrict__ C,
                           int M, int Ncols, int Kdim)
{
    __shared__ bf16 lA[TM][LPAD];   // A tile, row-major (row m, k)
    __shared__ bf16 lB[TN][LPAD];   // B tile, transposed (col n, k)

    const int tid  = threadIdx.x;
    const int lane = tid & 31;
    const int wave = tid >> 5;
    const int rowBase = blockIdx.y * TM;
    const int colBase = blockIdx.x * TN;

    v8f acc[4];
#pragma unroll
    for (int i = 0; i < 4; ++i)
#pragma unroll
        for (int r = 0; r < 8; ++r)
            acc[i][r] = 0.0f;

    // ISA fragment coordinates for this lane
    const int mA  = lane & 15;            // A row within 16
    const int kbA = (lane >> 4) * 8;      // A: lanes 0-15 K base 0, lanes 16-31 K base 8
    const int nB  = lane & 15;            // B column within 16
    const int kbB = (lane >> 4) * 8;      // B: lanes 0-15 K base 0, lanes 16-31 K base 8

    for (int kk = 0; kk < Kdim; kk += TK) {
        __syncthreads();
        // ---- Stage A tile (TM x TK) via async memory->LDS copies (16B each).
        // Out-of-range rows are clamped to M-1: they only feed D rows >= M,
        // which are never stored, so results are unaffected and EXEC stays
        // uniform (no divergent guard around the async op).
#pragma unroll
        for (int i = tid; i < (TM * TK) / 8; i += 128) {
            int r  = i >> 2;
            int cc = (i & 3) * 8;
            int gr = rowBase + r;
            if (gr >= M) gr = M - 1;
            unsigned ldsOff = (unsigned)(size_t)&lA[r][cc];
            unsigned long long gaddr =
                (unsigned long long)(A + (size_t)gr * Kdim + kk + cc);
            asm volatile("global_load_async_to_lds_b128 %0, %1, off"
                         :: "v"(ldsOff), "v"(gaddr) : "memory");
        }
        // ---- Stage B tile (TK x TN), transposed into lB[col][k]
#pragma unroll
        for (int i = tid; i < TK * TN; i += 128) {
            int kr = i >> 6;
            int cc = i & 63;
            lB[cc][kr] = B[(size_t)(kk + kr) * Ncols + colBase + cc];
        }
        asm volatile("s_wait_asynccnt 0x0" ::: "memory");  // A tile landed in LDS
        __syncthreads();                                   // + B tile visible

        // A fragment: elems 0..7 -> K = kbA..kbA+7, elems 8..15 -> K = 16+kbA..
        FragBF a;
        a.u[0] = *(const u32x4*)(&lA[wave * 16 + mA][kbA]);
        a.u[1] = *(const u32x4*)(&lA[wave * 16 + mA][16 + kbA]);
        // Load all four B fragments first so the 4 WMMAs can issue back-to-back
        FragBF b[4];
#pragma unroll
        for (int cb = 0; cb < 4; ++cb) {
            b[cb].u[0] = *(const u32x4*)(&lB[cb * 16 + nB][kbB]);
            b[cb].u[1] = *(const u32x4*)(&lB[cb * 16 + nB][16 + kbB]);
        }
#pragma unroll
        for (int cb = 0; cb < 4; ++cb) {
            acc[cb] = __builtin_amdgcn_wmma_f32_16x16x32_bf16(
                false, a.v, false, b[cb].v, (short)0, acc[cb], false, false);
        }
    }

    // C layout: VGPR r -> row r (lanes 0-15) / row 8+r (lanes 16-31), col = lane&15
    const int mHalf = (lane >> 4) * 8;
#pragma unroll
    for (int cb = 0; cb < 4; ++cb) {
        int col = colBase + cb * 16 + nB;
        float bv = bias ? bias[col] : 0.0f;
#pragma unroll
        for (int r = 0; r < 8; ++r) {
            int row = rowBase + wave * 16 + mHalf + r;
            if (row < M)
                C[(size_t)row * Ncols + col] = acc[cb][r] + bv;
        }
    }
}

// ---------------------------------------------------------------------------
// Per-edge Gaussian mixture weights: gw[e,k] = exp(sum_d -0.5 (e_d-mu_kd)^2/sig^2)
// ---------------------------------------------------------------------------
__global__ void edge_gauss_kernel(const float* __restrict__ eattr,
                                  const float* __restrict__ mu,
                                  const float* __restrict__ sig,
                                  float* __restrict__ gw, int E, int KK)
{
    int e = blockIdx.x * blockDim.x + threadIdx.x;
    if (e >= E) return;
    float a0 = eattr[e * 3 + 0];
    float a1 = eattr[e * 3 + 1];
    float a2 = eattr[e * 3 + 2];
    for (int k = 0; k < KK; ++k) {
        float s = 0.0f;
        float d0 = a0 - mu[k * 3 + 0], s0 = sig[k * 3 + 0];
        float d1 = a1 - mu[k * 3 + 1], s1 = sig[k * 3 + 1];
        float d2 = a2 - mu[k * 3 + 2], s2 = sig[k * 3 + 2];
        s -= 0.5f * d0 * d0 / (1e-15f + s0 * s0);
        s -= 0.5f * d1 * d1 / (1e-15f + s1 * s1);
        s -= 0.5f * d2 * d2 / (1e-15f + s2 * s2);
        gw[e * KK + k] = expf(s);
    }
}

// ---------------------------------------------------------------------------
// Edge message + scatter: one thread per (edge, out-channel).
// Gathers Y[src] (L2-resident on MI455X: Y=128MB < 192MB L2), atomic-adds to
// agg[dst]. cnt[dst] incremented once per edge.
// ---------------------------------------------------------------------------
__global__ void edge_scatter_kernel(const int* __restrict__ ei,
                                    const float* __restrict__ gw,
                                    const float* __restrict__ Y,
                                    float* __restrict__ agg,
                                    float* __restrict__ cnt,
                                    int E, int KK)
{
    int idx = blockIdx.x * blockDim.x + threadIdx.x;
    int e = idx >> 7;               // COUT == 128
    if (e >= E) return;
    int c = idx & (COUT - 1);
    int src = ei[e];
    int dst = ei[E + e];
    const float* yrow = Y + (size_t)src * (size_t)(KK * COUT) + c;
    const float* gwe  = gw + (size_t)e * KK;
    __builtin_prefetch(yrow, 0, 0);   // global_prefetch_b8
    float m = 0.0f;
    for (int k = 0; k < KK; ++k)
        m += gwe[k] * yrow[(size_t)k * COUT];
    __hip_atomic_fetch_add(&agg[(size_t)dst * COUT + c], m,
                           __ATOMIC_RELAXED, __HIP_MEMORY_SCOPE_AGENT);
    if (c == 0)
        __hip_atomic_fetch_add(&cnt[dst], 1.0f,
                               __ATOMIC_RELAXED, __HIP_MEMORY_SCOPE_AGENT);
}

// ---------------------------------------------------------------------------
// Mean-aggregate + root term: H = (cnt>0 ? agg/cnt : 0) + R
// ---------------------------------------------------------------------------
__global__ void combine_kernel(const float* __restrict__ agg,
                               const float* __restrict__ cnt,
                               const float* __restrict__ R,
                               float* __restrict__ H, int total)
{
    int i = blockIdx.x * blockDim.x + threadIdx.x;
    if (i >= total) return;
    int n = i >> 7;
    float h = R[i];
    float ct = cnt[n];
    if (ct > 0.0f) h += agg[i] / ct;
    H[i] = h;
}

// ---------------------------------------------------------------------------
// BatchNorm batch statistics: one block per channel, LDS tree reduce.
// meanrstd[c] = mean, meanrstd[C + c] = rsqrt(var + 1e-5)
// ---------------------------------------------------------------------------
__global__ void bn_stats_kernel(const float* __restrict__ H, int Nrows, int C,
                                float* __restrict__ meanrstd)
{
    int c = blockIdx.x;
    int tid = threadIdx.x;
    float s = 0.0f, q = 0.0f;
    for (int n = tid; n < Nrows; n += blockDim.x) {
        float v = H[(size_t)n * C + c];
        s += v; q += v * v;
    }
    __shared__ float ss[256];
    __shared__ float sq[256];
    ss[tid] = s; sq[tid] = q;
    __syncthreads();
    for (int off = 128; off > 0; off >>= 1) {
        if (tid < off) { ss[tid] += ss[tid + off]; sq[tid] += sq[tid + off]; }
        __syncthreads();
    }
    if (tid == 0) {
        float m   = ss[0] / (float)Nrows;
        float var = sq[0] / (float)Nrows - m * m;   // biased variance
        meanrstd[c]     = m;
        meanrstd[C + c] = rsqrtf(var + 1e-5f);
    }
}

// ---------------------------------------------------------------------------
// BN + ELU + convert to bf16 (conv1 output -> conv2 input)
// ---------------------------------------------------------------------------
__global__ void bn_elu_cvt_kernel(const float* __restrict__ H,
                                  const float* __restrict__ mr,
                                  const float* __restrict__ gamma,
                                  const float* __restrict__ beta,
                                  bf16* __restrict__ out, int total)
{
    int i = blockIdx.x * blockDim.x + threadIdx.x;
    if (i >= total) return;
    int c = i & (COUT - 1);
    float v = (H[i] - mr[c]) * mr[COUT + c] * gamma[c] + beta[c];
    v = v > 0.0f ? v : expm1f(v);
    out[i] = (bf16)v;
}

// ---------------------------------------------------------------------------
// Final: elu( bn(H2) + bn(SC) ) -> f32 output
// ---------------------------------------------------------------------------
__global__ void final_kernel(const float* __restrict__ H2,
                             const float* __restrict__ SC,
                             const float* __restrict__ mr2,
                             const float* __restrict__ mr3,
                             const float* __restrict__ gam2,
                             const float* __restrict__ bet2,
                             const float* __restrict__ gams,
                             const float* __restrict__ bets,
                             float* __restrict__ out, int total)
{
    int i = blockIdx.x * blockDim.x + threadIdx.x;
    if (i >= total) return;
    int c = i & (COUT - 1);
    float a = (H2[i] - mr2[c]) * mr2[COUT + c] * gam2[c] + bet2[c];
    float b = (SC[i] - mr3[c]) * mr3[COUT + c] * gams[c] + bets[c];
    float v = a + b;
    out[i] = v > 0.0f ? v : expm1f(v);
}

// ---------------------------------------------------------------------------
// Host-side orchestration of one GMM conv layer
// ---------------------------------------------------------------------------
static void launch_conv(const bf16* Xbf, int Cin,
                        const float* g, const float* mu, const float* sig,
                        const float* root, const float* bias, int KK,
                        const int* ei, const float* eattr,
                        bf16* wbf, bf16* rbf, float* Y, float* R,
                        float* agg, float* cnt, float* gw,
                        float* Hout, float* meanrstd, hipStream_t stream)
{
    const int Nc = KK * COUT;
    cvt_bf16_kernel<<<ceil_div(Cin * Nc, 256), 256, 0, stream>>>(g, wbf, Cin * Nc);
    cvt_bf16_kernel<<<ceil_div(Cin * COUT, 256), 256, 0, stream>>>(root, rbf, Cin * COUT);

    dim3 gy(Nc / TN, ceil_div(NNODES, TM));
    wmma_gemm_bf16_kernel<<<gy, 128, 0, stream>>>(Xbf, wbf, nullptr, Y, NNODES, Nc, Cin);
    dim3 gr(COUT / TN, ceil_div(NNODES, TM));
    wmma_gemm_bf16_kernel<<<gr, 128, 0, stream>>>(Xbf, rbf, bias, R, NNODES, COUT, Cin);

    edge_gauss_kernel<<<ceil_div(NEDGES, 256), 256, 0, stream>>>(eattr, mu, sig, gw, NEDGES, KK);
    fill_zero_kernel<<<ceil_div(NNODES * COUT, 256), 256, 0, stream>>>(agg, NNODES * COUT);
    fill_zero_kernel<<<ceil_div(NNODES, 256), 256, 0, stream>>>(cnt, NNODES);
    edge_scatter_kernel<<<ceil_div(NEDGES * COUT, 256), 256, 0, stream>>>(
        ei, gw, Y, agg, cnt, NEDGES, KK);
    combine_kernel<<<ceil_div(NNODES * COUT, 256), 256, 0, stream>>>(
        agg, cnt, R, Hout, NNODES * COUT);
    bn_stats_kernel<<<COUT, 256, 0, stream>>>(Hout, NNODES, COUT, meanrstd);
}

extern "C" void kernel_launch(void* const* d_in, const int* in_sizes, int n_in,
                              void* d_out, int out_size, void* d_ws, size_t ws_size,
                              hipStream_t stream)
{
    (void)in_sizes; (void)n_in; (void)out_size; (void)ws_size;

    const float* x     = (const float*)d_in[0];
    const int*   ei    = (const int*)d_in[1];
    const float* eattr = (const float*)d_in[2];
    const float* g1    = (const float*)d_in[3];
    const float* mu1   = (const float*)d_in[4];
    const float* sig1  = (const float*)d_in[5];
    const float* root1 = (const float*)d_in[6];
    const float* b1    = (const float*)d_in[7];
    const float* gam1  = (const float*)d_in[8];
    const float* bet1  = (const float*)d_in[9];
    const float* g2    = (const float*)d_in[10];
    const float* mu2   = (const float*)d_in[11];
    const float* sig2  = (const float*)d_in[12];
    const float* root2 = (const float*)d_in[13];
    const float* b2    = (const float*)d_in[14];
    const float* gam2  = (const float*)d_in[15];
    const float* bet2  = (const float*)d_in[16];
    const float* gs    = (const float*)d_in[17];
    const float* mus   = (const float*)d_in[18];
    const float* sigs  = (const float*)d_in[19];
    const float* roots = (const float*)d_in[20];
    const float* bs    = (const float*)d_in[21];
    const float* gams  = (const float*)d_in[22];
    const float* bets  = (const float*)d_in[23];

    // Workspace carve-out (256B aligned)
    char* ws = (char*)d_ws;
    size_t off = 0;
    auto alloc = [&](size_t bytes) -> char* {
        char* p = ws + off;
        off += (bytes + 255) & ~(size_t)255;
        return p;
    };
    bf16*  xbf = (bf16*)alloc((size_t)NNODES * CIN * 2);          //  6.4 MB
    bf16*  hbf = (bf16*)alloc((size_t)NNODES * COUT * 2);         // 12.8 MB
    bf16*  wbf = (bf16*)alloc((size_t)COUT * KMIX * COUT * 2);    //  164 KB
    bf16*  rbf = (bf16*)alloc((size_t)COUT * COUT * 2);           //   33 KB
    float* Y   = (float*)alloc((size_t)NNODES * KMIX * COUT * 4); //  128 MB (reused x3)
    float* R   = (float*)alloc((size_t)NNODES * COUT * 4);        // 25.6 MB (reused)
    float* agg = (float*)alloc((size_t)NNODES * COUT * 4);        // 25.6 MB (reused)
    float* cnt = (float*)alloc((size_t)NNODES * 4);               //  0.2 MB (reused)
    float* gw  = (float*)alloc((size_t)NEDGES * KMIX * 4);        //   12 MB (reused)
    float* HA  = (float*)alloc((size_t)NNODES * COUT * 4);        // conv1 out, then conv2 out
    float* HB  = (float*)alloc((size_t)NNODES * COUT * 4);        // shortcut out
    float* mr1 = (float*)alloc((size_t)2 * COUT * 4);
    float* mr2 = (float*)alloc((size_t)2 * COUT * 4);
    float* mr3 = (float*)alloc((size_t)2 * COUT * 4);

    const int total = NNODES * COUT;

    // ---- conv1: h = elu(bn(gmm_conv(x, g1...))) ----
    cvt_bf16_kernel<<<ceil_div(NNODES * CIN, 256), 256, 0, stream>>>(x, xbf, NNODES * CIN);
    launch_conv(xbf, CIN, g1, mu1, sig1, root1, b1, KMIX, ei, eattr,
                wbf, rbf, Y, R, agg, cnt, gw, HA, mr1, stream);
    bn_elu_cvt_kernel<<<ceil_div(total, 256), 256, 0, stream>>>(HA, mr1, gam1, bet1, hbf, total);

    // ---- conv2: bn(gmm_conv(h, g2...)) -> HA (reuse; H1 no longer needed) ----
    launch_conv(hbf, COUT, g2, mu2, sig2, root2, b2, KMIX, ei, eattr,
                wbf, rbf, Y, R, agg, cnt, gw, HA, mr2, stream);

    // ---- shortcut: bn(gmm_conv(x, gs...)) -> HB (K=1) ----
    launch_conv(xbf, CIN, gs, mus, sigs, roots, bs, 1, ei, eattr,
                wbf, rbf, Y, R, agg, cnt, gw, HB, mr3, stream);

    // ---- final: elu(bn2 + bn3) ----
    final_kernel<<<ceil_div(total, 256), 256, 0, stream>>>(
        HA, HB, mr2, mr3, gam2, bet2, gams, bets, (float*)d_out, total);
}
